// NewsEncoder_89919435309481
// MI455X (gfx1250) — matrix operand order repaired and checked by reading, not compiled, not run on hardware
//
#include <hip/hip_runtime.h>
#include <hip/hip_bf16.h>
#include <math.h>

typedef __attribute__((ext_vector_type(16))) _Float16 v16h;
typedef __attribute__((ext_vector_type(8)))  float    v8f;
typedef __attribute__((ext_vector_type(4)))  int      v4i;
typedef v4i __attribute__((address_space(1))) g_v4i;   // global 128-bit chunk
typedef v4i __attribute__((address_space(3))) l_v4i;   // LDS 128-bit chunk

#define B_   512
#define T_   720
#define D_   774
#define DA_  776        // D + has + rec
#define KP_  800        // DA padded to 25*32 for WMMA K-steps
#define P_   128
#define H_   128
#define G_   512        // 4*H
#define E_   64
#define BT_  (B_*T_)
#define BM_  16         // LSTM batch rows per workgroup

__device__ __forceinline__ float gelu_exact(float v) {
    return 0.5f * v * (1.0f + erff(v * 0.70710678118654752f));
}
__device__ __forceinline__ float sigmoidf_(float v) {
    return 1.0f / (1.0f + __expf(-v));
}

// ---- CDNA5 async global->LDS copy (16B per lane), tracked by ASYNCcnt ------
__device__ __forceinline__ void async_cp16(const void* g, void* l) {
#if __has_builtin(__builtin_amdgcn_global_load_async_to_lds_b128)
    __builtin_amdgcn_global_load_async_to_lds_b128(
        (g_v4i*)(void*)g, (l_v4i*)l, 0, 0);
#else
    asm volatile("global_load_async_to_lds_b128 %0, %1, off"
                 : : "v"((unsigned)(unsigned long long)l), "v"(g) : "memory");
#endif
}
__device__ __forceinline__ void wait_async0() {
#if __has_builtin(__builtin_amdgcn_s_wait_asynccnt)
    __builtin_amdgcn_s_wait_asynccnt(0);
#else
    asm volatile("s_wait_asynccnt 0x0" : : : "memory");
#endif
}

// ---- WMMA fragment helpers (CDNA5 16x16x32 f16, wave32) --------------------
__device__ __forceinline__ v16h load_a_frag(const _Float16* base, int row0, int k0, int ld) {
    const int lane = threadIdx.x & 31;
    const int row  = row0 + (lane & 15);
    const int half = lane >> 4;
    const _Float16* p = base + row * ld + k0 + half * 16;
    v16h a;
#pragma unroll
    for (int j = 0; j < 8; ++j) a[j] = p[j];
#pragma unroll
    for (int j = 0; j < 8; ++j) a[8 + j] = p[8 + j];
    return a;
}
// B from TRANSPOSED storage [N][K]: one contiguous 16-half (32B) run per lane.
__device__ __forceinline__ v16h load_b_frag_t(const _Float16* base, int col0, int k0, int ldk) {
    const int lane = threadIdx.x & 31;
    const int col  = col0 + (lane & 15);
    const int half = lane >> 4;
    const _Float16* p = base + col * ldk + k0 + half * 16;
    v16h b;
#pragma unroll
    for (int j = 0; j < 16; ++j) b[j] = p[j];
    return b;
}
__device__ __forceinline__ v8f wmma16(v16h a, v16h b, v8f c) {
    return __builtin_amdgcn_wmma_f32_16x16x32_f16(false, a, false, b, (short)0, c, false, false);
}
// C/D tile: lane L, vgpr r -> row = r + 8*(L>>4), col = 16*n + (L&15)

// ---------------------------------------------------------------------------
__global__ void has_kernel(const float* __restrict__ x, float* __restrict__ has) {
    const int wave = threadIdx.x >> 5;
    const int lane = threadIdx.x & 31;
    const long r = (long)blockIdx.x * 8 + wave;
    const float* xr = x + r * (long)D_;
    float s = 0.f;
    for (int c = lane; c < D_; c += 32) s += fabsf(xr[c]);
#pragma unroll
    for (int off = 16; off > 0; off >>= 1) s += __shfl_xor(s, off, 32);
    if (lane == 0) has[r] = (s > 1e-6f) ? 1.0f : 0.0f;
}

__global__ void rec_kernel(const float* __restrict__ has, float* __restrict__ rec) {
    if (threadIdx.x != 0) return;
    const int b = blockIdx.x;
    float last = -1.0f;
    for (int t = 0; t < T_; ++t) {
        const long i = (long)b * T_ + t;
        if (has[i] > 0.5f) last = (float)t;
        float rv = ((float)t - last) / (float)T_;
        rv = fminf(fmaxf(rv, 0.0f), 1.0f);
        rec[i] = rv;
    }
}

// weight prep: transpose + f16 convert.  dst[n][k] = src[k][n]
__global__ void transpose_cvt_kernel(const float* __restrict__ s, _Float16* __restrict__ d,
                                     int K, int N) {
    int i = blockIdx.x * blockDim.x + threadIdx.x;
    if (i >= K * N) return;
    int n = i / K, k = i % K;
    d[i] = (_Float16)s[k * N + n];
}
__global__ void wprojT_kernel(const float* __restrict__ s, _Float16* __restrict__ d) {
    int i = blockIdx.x * blockDim.x + threadIdx.x;
    if (i >= P_ * KP_) return;
    int n = i / KP_, k = i % KP_;
    d[i] = (k < DA_) ? (_Float16)s[k * P_ + n] : (_Float16)0.0f;
}
__global__ void bias_add_kernel(const float* __restrict__ a, const float* __restrict__ b,
                                float* __restrict__ o, int n) {
    int i = blockIdx.x * blockDim.x + threadIdx.x;
    if (i < n) o[i] = a[i] + b[i];
}

// ---------------------------------------------------------------------------
// Projection GEMM + GELU + LayerNorm. Block: 128 rows x 128 cols, 8 waves.
// B^T slice arrives via async copy, overlapped with VALU A-conversion.
struct __align__(16) ProjStage { _Float16 As[128][40]; _Float16 BsT[128][40]; };
union  __align__(16) ProjLDS   { ProjStage st; float Es[128][128]; };

__global__ void __launch_bounds__(256)
proj_kernel(const float* __restrict__ x, const float* __restrict__ has,
            const float* __restrict__ rec, const _Float16* __restrict__ wprojT,
            const float* __restrict__ bproj, const float* __restrict__ lng,
            const float* __restrict__ lnb, _Float16* __restrict__ h0) {
    __shared__ ProjLDS L;
    const int tid  = threadIdx.x;
    const int wave = tid >> 5;
    const int lane = tid & 31;
    const int half = lane >> 4;
    const long rowBase = (long)blockIdx.x * 128;

    v8f acc[8];
#pragma unroll
    for (int n = 0; n < 8; ++n) acc[n] = (v8f)(0.0f);

    for (int kk = 0; kk < KP_ / 32; ++kk) {
        __syncthreads();
        {   // async B^T slice: 128 rows x 64B -> 512 chunks, 2/thread
#pragma unroll
            for (int i = 0; i < 2; ++i) {
                const int c = tid + 256 * i;
                const int row = c >> 2, o = (c & 3) * 8;
                async_cp16(wprojT + row * KP_ + kk * 32 + o, &L.st.BsT[row][o]);
            }
        }
        {   // A: 128x32 from f32 x (+has,rec,pad) -> f16  (VALU, overlaps async)
            const int row = tid >> 1, c0 = (tid & 1) * 16;
            const long gRow = rowBase + row;
#pragma unroll
            for (int j = 0; j < 16; ++j) {
                int col = kk * 32 + c0 + j;
                float v;
                if (col < D_)            v = x[gRow * (long)D_ + col];
                else if (col == D_)      v = has[gRow];
                else if (col == D_ + 1)  v = rec[gRow];
                else                     v = 0.0f;
                L.st.As[row][c0 + j] = (_Float16)v;
            }
        }
        wait_async0();
        __syncthreads();
        v16h a = load_a_frag(&L.st.As[0][0], 16 * wave, 0, 40);
#pragma unroll
        for (int n = 0; n < 8; ++n) {
            v16h b = load_b_frag_t(&L.st.BsT[0][0], 16 * n, 0, 40);
            acc[n] = wmma16(a, b, acc[n]);
        }
    }
    __syncthreads();
#pragma unroll
    for (int n = 0; n < 8; ++n) {
        const int col = 16 * n + (lane & 15);
        const float bb = bproj[col];
#pragma unroll
        for (int r = 0; r < 8; ++r) {
            int row = 16 * wave + r + 8 * half;
            L.Es[row][col] = gelu_exact(acc[n][r] + bb);
        }
    }
    __syncthreads();
    if (tid < 128) {  // LayerNorm per row
        const int row = tid;
        float s = 0.f, s2 = 0.f;
#pragma unroll 8
        for (int c = 0; c < 128; ++c) { float v = L.Es[row][c]; s += v; s2 += v * v; }
        const float mu = s * (1.0f / 128.0f);
        const float var = s2 * (1.0f / 128.0f) - mu * mu;
        const float rstd = rsqrtf(var + 1e-5f);
        const long gRow = rowBase + row;
#pragma unroll 8
        for (int c = 0; c < 128; ++c) {
            float v = (L.Es[row][c] - mu) * rstd * lng[c] + lnb[c];
            h0[gRow * P_ + c] = (_Float16)v;
        }
    }
}

// ---------------------------------------------------------------------------
// Input-side GEMM: out[M,512] = A[M,128] @ W + bias (W as W^T[512][128]).
// B^T resident (async once); A K-slices double-buffered via async copies.
__global__ void __launch_bounds__(256)
pre_gemm_kernel(const _Float16* __restrict__ A, const _Float16* __restrict__ WT,
                const float* __restrict__ bias, _Float16* __restrict__ out) {
    __shared__ __align__(16) _Float16 BsT[128][136];   // 34 KB
    __shared__ __align__(16) _Float16 As[2][128][40];  // 2 x 10 KB
    const int tid  = threadIdx.x;
    const int wave = tid >> 5;
    const int lane = tid & 31;
    const int half = lane >> 4;
    const long rowBase = (long)blockIdx.x * 128;
    const int  colBase = blockIdx.y * 128;

    // resident B^T block: 128 rows x 256B -> 2048 chunks, 8/thread
#pragma unroll
    for (int i = 0; i < 8; ++i) {
        const int c = tid + 256 * i;
        const int row = c >> 4, o = (c & 15) * 8;
        async_cp16(WT + (long)(colBase + row) * P_ + o, &BsT[row][o]);
    }
    // A slice 0: 128 rows x 64B -> 512 chunks, 2/thread
#pragma unroll
    for (int i = 0; i < 2; ++i) {
        const int c = tid + 256 * i;
        const int row = c >> 2, o = (c & 3) * 8;
        async_cp16(A + (rowBase + row) * (long)P_ + o, &As[0][row][o]);
    }
    wait_async0();
    __syncthreads();

    v8f acc[8];
#pragma unroll
    for (int n = 0; n < 8; ++n) acc[n] = (v8f)(0.0f);

#pragma unroll 1
    for (int kk = 0; kk < 4; ++kk) {
        const int cur = kk & 1;
        if (kk < 3) {  // prefetch next A slice into other buffer
#pragma unroll
            for (int i = 0; i < 2; ++i) {
                const int c = tid + 256 * i;
                const int row = c >> 2, o = (c & 3) * 8;
                async_cp16(A + (rowBase + row) * (long)P_ + (kk + 1) * 32 + o,
                           &As[cur ^ 1][row][o]);
            }
        }
        v16h a = load_a_frag(&As[cur][0][0], 16 * wave, 0, 40);
#pragma unroll
        for (int n = 0; n < 8; ++n) {
            v16h b = load_b_frag_t(&BsT[0][0], 16 * n, kk * 32, 136);
            acc[n] = wmma16(a, b, acc[n]);
        }
        wait_async0();
        __syncthreads();
    }
#pragma unroll
    for (int n = 0; n < 8; ++n) {
        const int col = colBase + 16 * n + (lane & 15);
        const float bb = bias[col];
#pragma unroll
        for (int r = 0; r < 8; ++r) {
            long gRow = rowBase + 16 * wave + r + 8 * half;
            out[gRow * (long)G_ + col] = (_Float16)(acc[n][r] + bb);
        }
    }
}

// ---------------------------------------------------------------------------
// LSTM recurrence: 16 batch rows/WG (grid=32), T steps in-kernel.
// Whh^T fragments register-resident; pre[:,t,:] tiles async double-buffered.
__global__ void __launch_bounds__(256, 1)
lstm_seq_kernel(const _Float16* __restrict__ pre, const _Float16* __restrict__ WhhT,
                _Float16* __restrict__ out_seq, float* __restrict__ last_h) {
    __shared__ __align__(16) _Float16 Hs[BM_][136];     // hidden state (padded)
    __shared__            float    Cs[BM_][128];        // cell state
    __shared__ __align__(16) _Float16 Gs[BM_][512];     // recurrent contributions
    __shared__ __align__(16) _Float16 Ps[2][BM_][512];  // async pre tiles (2x16KB)

    const int tid  = threadIdx.x;
    const int wave = tid >> 5;
    const int lane = tid & 31;
    const int half = lane >> 4;
    const int bBase = blockIdx.x * BM_;

    // Whh^T fragments once: 4 k-steps x 4 n-tiles = 128 VGPRs (weight-stationary)
    v16h bf[4][4];
#pragma unroll
    for (int kk = 0; kk < 4; ++kk)
#pragma unroll
        for (int nt = 0; nt < 4; ++nt)
            bf[kk][nt] = load_b_frag_t(WhhT, wave * 64 + 16 * nt, kk * 32, P_);

    // prefetch pre tile for t=0: 16 rows x 1024B -> 1024 chunks, 4/thread
#pragma unroll
    for (int i = 0; i < 4; ++i) {
        const int c = tid + 256 * i;
        const int b = c >> 6, o = (c & 63) * 8;
        async_cp16(pre + ((long)(bBase + b) * T_ + 0) * G_ + o, &Ps[0][b][o]);
    }
    for (int i = tid; i < BM_ * 136; i += 256) (&Hs[0][0])[i] = (_Float16)0.0f;
    for (int i = tid; i < BM_ * 128; i += 256) (&Cs[0][0])[i] = 0.0f;
    wait_async0();
    __syncthreads();

    for (int t = 0; t < T_; ++t) {
        const int cur = t & 1;
        if (t + 1 < T_) {  // prefetch next step's gate tile
#pragma unroll
            for (int i = 0; i < 4; ++i) {
                const int c = tid + 256 * i;
                const int b = c >> 6, o = (c & 63) * 8;
                async_cp16(pre + ((long)(bBase + b) * T_ + (t + 1)) * G_ + o,
                           &Ps[cur ^ 1][b][o]);
            }
        }
        v8f acc[4];
#pragma unroll
        for (int nt = 0; nt < 4; ++nt) acc[nt] = (v8f)(0.0f);
#pragma unroll
        for (int kk = 0; kk < 4; ++kk) {
            v16h a = load_a_frag(&Hs[0][0], 0, kk * 32, 136);
#pragma unroll
            for (int nt = 0; nt < 4; ++nt) acc[nt] = wmma16(a, bf[kk][nt], acc[nt]);
        }
#pragma unroll
        for (int nt = 0; nt < 4; ++nt) {
            const int col = wave * 64 + 16 * nt + (lane & 15);
#pragma unroll
            for (int r = 0; r < 8; ++r) Gs[r + 8 * half][col] = (_Float16)acc[nt][r];
        }
        __syncthreads();
        // gate math from LDS only
        for (int i = tid; i < BM_ * 128; i += 256) {
            const int b = i >> 7, j = i & 127;
            const long gb = bBase + b;
            float gi = (float)Gs[b][j]       + (float)Ps[cur][b][j];
            float gf = (float)Gs[b][128 + j] + (float)Ps[cur][b][128 + j];
            float gg = (float)Gs[b][256 + j] + (float)Ps[cur][b][256 + j];
            float go = (float)Gs[b][384 + j] + (float)Ps[cur][b][384 + j];
            float c = sigmoidf_(gf) * Cs[b][j] + sigmoidf_(gi) * tanhf(gg);
            float h = sigmoidf_(go) * tanhf(c);
            Cs[b][j] = c;
            Hs[b][j] = (_Float16)h;
            if (out_seq) out_seq[(gb * (long)T_ + t) * P_ + j] = (_Float16)h;
            if (last_h && t == T_ - 1) last_h[gb * H_ + j] = h;
        }
        wait_async0();
        __syncthreads();
    }
}

// ---------------------------------------------------------------------------
__global__ void __launch_bounds__(128)
heads_kernel(const float* __restrict__ lastH,
             const float* __restrict__ we1, const float* __restrict__ be1,
             const float* __restrict__ we2, const float* __restrict__ be2,
             const float* __restrict__ wr,  const float* __restrict__ br,
             const float* __restrict__ wo,  const float* __restrict__ bo,
             const float* __restrict__ ws,
             float* __restrict__ out_pred, float* __restrict__ out_emb) {
    __shared__ float lh[128], e1s[128], hrs[64], red[128];
    const int b = blockIdx.x, tid = threadIdx.x;
    lh[tid] = lastH[b * H_ + tid];
    __syncthreads();
    {
        float s = be1[tid];
#pragma unroll 8
        for (int k = 0; k < 128; ++k) s += lh[k] * we1[k * H_ + tid];
        e1s[tid] = gelu_exact(s);
    }
    if (tid < 64) {
        float s = br[tid];
#pragma unroll 8
        for (int k = 0; k < 128; ++k) s += lh[k] * wr[k * (H_ / 2) + tid];
        hrs[tid] = gelu_exact(s);
    }
    __syncthreads();
    if (tid < 64) {
        float s = be2[tid];
#pragma unroll 8
        for (int k = 0; k < 128; ++k) s += e1s[k] * we2[k * E_ + tid];
        out_emb[b * E_ + tid] = tanhf(s);
    }
    float p = lh[tid] * ws[tid];
    if (tid < 64) p += hrs[tid] * wo[tid];
    red[tid] = p;
    __syncthreads();
#pragma unroll
    for (int off = 64; off > 0; off >>= 1) {
        if (tid < off) red[tid] += red[tid + off];
        __syncthreads();
    }
    if (tid == 0) out_pred[b] = red[0] + bo[0];
}

// ---------------------------------------------------------------------------
extern "C" void kernel_launch(void* const* d_in, const int* in_sizes, int n_in,
                              void* d_out, int out_size, void* d_ws, size_t ws_size,
                              hipStream_t stream) {
    (void)in_sizes; (void)n_in; (void)out_size; (void)ws_size;
    const float* x      = (const float*)d_in[0];
    const float* w_proj = (const float*)d_in[1];
    const float* b_proj = (const float*)d_in[2];
    const float* ln_g   = (const float*)d_in[3];
    const float* ln_b   = (const float*)d_in[4];
    const float* wih0   = (const float*)d_in[5];
    const float* whh0   = (const float*)d_in[6];
    const float* bih0   = (const float*)d_in[7];
    const float* bhh0   = (const float*)d_in[8];
    const float* wih1   = (const float*)d_in[9];
    const float* whh1   = (const float*)d_in[10];
    const float* bih1   = (const float*)d_in[11];
    const float* bhh1   = (const float*)d_in[12];
    const float* w_e1   = (const float*)d_in[13];
    const float* b_e1   = (const float*)d_in[14];
    const float* w_e2   = (const float*)d_in[15];
    const float* b_e2   = (const float*)d_in[16];
    const float* w_r    = (const float*)d_in[17];
    const float* b_r    = (const float*)d_in[18];
    const float* w_o    = (const float*)d_in[19];
    const float* b_o    = (const float*)d_in[20];
    const float* w_s    = (const float*)d_in[21];

    char* w = (char*)d_ws;
    size_t off = 0;
    auto nxt = [&](size_t bytes) -> char* {
        char* p = w + off;
        off = (off + bytes + 255) & ~(size_t)255;
        return p;
    };
    float*     has     = (float*)nxt((size_t)BT_ * 4);
    float*     rec     = (float*)nxt((size_t)BT_ * 4);
    _Float16*  wprojT  = (_Float16*)nxt((size_t)P_ * KP_ * 2);
    _Float16*  wih0T   = (_Float16*)nxt((size_t)G_ * P_ * 2);
    _Float16*  whh0T   = (_Float16*)nxt((size_t)G_ * H_ * 2);
    _Float16*  wih1T   = (_Float16*)nxt((size_t)G_ * H_ * 2);
    _Float16*  whh1T   = (_Float16*)nxt((size_t)G_ * H_ * 2);
    float*     bias0   = (float*)nxt((size_t)G_ * 4);
    float*     bias1   = (float*)nxt((size_t)G_ * 4);
    _Float16*  h0      = (_Float16*)nxt((size_t)BT_ * P_ * 2);
    _Float16*  hs0     = (_Float16*)nxt((size_t)BT_ * P_ * 2);
    _Float16*  pre     = (_Float16*)nxt((size_t)BT_ * G_ * 2);
    float*     lasth   = (float*)nxt((size_t)B_ * H_ * 4);

    float* out_pred = (float*)d_out;
    float* out_emb  = out_pred + B_;

    has_kernel<<<BT_ / 8, 256, 0, stream>>>(x, has);
    rec_kernel<<<B_, 32, 0, stream>>>(has, rec);

    transpose_cvt_kernel<<<(P_ * G_ + 255) / 256, 256, 0, stream>>>(wih0, wih0T, P_, G_);
    transpose_cvt_kernel<<<(H_ * G_ + 255) / 256, 256, 0, stream>>>(whh0, whh0T, H_, G_);
    transpose_cvt_kernel<<<(P_ * G_ + 255) / 256, 256, 0, stream>>>(wih1, wih1T, P_, G_);
    transpose_cvt_kernel<<<(H_ * G_ + 255) / 256, 256, 0, stream>>>(whh1, whh1T, H_, G_);
    wprojT_kernel<<<(P_ * KP_ + 255) / 256, 256, 0, stream>>>(w_proj, wprojT);
    bias_add_kernel<<<2, 256, 0, stream>>>(bih0, bhh0, bias0, G_);
    bias_add_kernel<<<2, 256, 0, stream>>>(bih1, bhh1, bias1, G_);

    proj_kernel<<<BT_ / 128, 256, 0, stream>>>(x, has, rec, wprojT, b_proj, ln_g, ln_b, h0);

    pre_gemm_kernel<<<dim3(BT_ / 128, 4), 256, 0, stream>>>(h0, wih0T, bias0, pre);
    lstm_seq_kernel<<<B_ / BM_, 256, 0, stream>>>(pre, whh0T, hs0, nullptr);

    pre_gemm_kernel<<<dim3(BT_ / 128, 4), 256, 0, stream>>>(hs0, wih1T, bias1, pre);
    lstm_seq_kernel<<<B_ / BM_, 256, 0, stream>>>(pre, whh1T, nullptr, lasth);

    heads_kernel<<<B_, 128, 0, stream>>>(lasth, w_e1, b_e1, w_e2, b_e2,
                                         w_r, b_r, w_o, b_o, w_s, out_pred, out_emb);
}
